// SwinTransformerLayer3D_14731737825299
// MI455X (gfx1250) — matrix operand order repaired and checked by reading, not compiled, not run on hardware
//
#include <hip/hip_runtime.h>
#include <hip/hip_bf16.h>

// ---------------------------------------------------------------------------
// Swin Transformer 3D layer for MI455X (gfx1250, wave32, WMMA 16x16x32 f16).
// GEMMs on v_wmma_f32_16x16x32_f16; A-operands staged to LDS with
// global_load_async_to_lds_b128 (ASYNCcnt) and read back via DS.
// ---------------------------------------------------------------------------

typedef __attribute__((ext_vector_type(16))) _Float16 v16h;
typedef __attribute__((ext_vector_type(8)))  float    v8f;

#define WMMA16x16x32(a, b, c) \
  __builtin_amdgcn_wmma_f32_16x16x32_f16(false, (a), false, (b), (short)0, (c), false, false)

namespace cfg {
constexpr int NH = 8;
constexpr int N  = 98;                 // tokens per window (2*7*7)
constexpr int NWB = 576;               // windows per batch (4*12*12)
constexpr int BW = 1152;               // total windows
constexpr int T  = 112896;             // total tokens  (= BW*N = B*D*H*W)
constexpr int MT = T / 16;             // 7056 M-tiles
constexpr float SCALE = 0.17677669529663687f;  // 1/sqrt(32)
}

// ------------------- async global -> LDS staging (CDNA5) -------------------
// Copies BYTES contiguous bytes (multiple of 16) from global to LDS using the
// gfx1250 async-DMA path; finishes with s_wait_asynccnt + workgroup barrier.
// Must be reached by all 256 threads of the block.
__device__ __forceinline__ void async_copy_to_lds(const void* gsrc, void* lds,
                                                  int bytes) {
  const int t = threadIdx.x;
  const unsigned lbase = (unsigned)(size_t)lds;   // low 32 bits = LDS offset
  const char* g = (const char*)gsrc;
  for (int off = t * 16; off < bytes; off += 256 * 16) {
    unsigned laddr = lbase + off;
    const char* gaddr = g + off;
    asm volatile("global_load_async_to_lds_b128 %0, %1, off"
                 :: "v"(laddr), "v"(gaddr)
                 : "memory");
  }
  asm volatile("s_wait_asynccnt 0" ::: "memory");
  __syncthreads();
}

// --------------------------- WMMA tile loaders -----------------------------
// A-matrix (16x32 f16), row-major src with leading dim ld (halves).
// Contiguous K order here (not ISA table order): lane L<16 -> row M=L, K=0..15; lane L>=16 -> row M=L-16,
// K=16..31.
__device__ __forceinline__ v16h load_a_tile(const _Float16* base, int ld) {
  const int lane = threadIdx.x & 31;
  const int row  = lane & 15;
  const int kb   = (lane < 16) ? 0 : 16;
  const _Float16* p = base + (size_t)row * ld;
  v16h a;
#pragma unroll
  for (int i = 0; i < 8; ++i) a[i] = p[kb + i];
#pragma unroll
  for (int i = 0; i < 8; ++i) a[8 + i] = p[kb + 8 + i];
  return a;
}

__device__ __forceinline__ v16h load_a_tile_cl(const _Float16* base, int ld,
                                               int row0, int rowmax) {
  const int lane = threadIdx.x & 31;
  int row = row0 + (lane & 15);
  if (row > rowmax) row = rowmax;
  const int kb = (lane < 16) ? 0 : 16;
  const _Float16* p = base + (size_t)row * ld;
  v16h a;
#pragma unroll
  for (int i = 0; i < 8; ++i) a[i] = p[kb + i];
#pragma unroll
  for (int i = 0; i < 8; ++i) a[8 + i] = p[kb + 8 + i];
  return a;
}

// B-matrix (32x16 f16) as transpose of row-major "Wrow": B[k][col]=Wrow[col][k].
// Contiguous K order here (same as A, not ISA table): lanes 0-15 hold K=0..15, lanes 16-31 hold K=16..31; N = lane&15.
__device__ __forceinline__ v16h load_bT_tile(const _Float16* base, int ld) {
  const int lane = threadIdx.x & 31;
  const int col  = lane & 15;
  const int kb   = (lane < 16) ? 0 : 16;
  const _Float16* p = base + (size_t)col * ld + kb;
  v16h b;
#pragma unroll
  for (int i = 0; i < 16; ++i) b[i] = p[i];
  return b;
}

__device__ __forceinline__ v16h load_bT_tile_cl(const _Float16* base, int ld,
                                                int col0, int colmax) {
  const int lane = threadIdx.x & 31;
  int col = col0 + (lane & 15);
  if (col > colmax) col = colmax;
  const int kb = (lane < 16) ? 0 : 16;
  const _Float16* p = base + (size_t)col * ld + kb;
  v16h b;
#pragma unroll
  for (int i = 0; i < 16; ++i) b[i] = p[i];
  return b;
}

// 256-thread block sum (8 wave32's), via shfl + 8-slot LDS exchange.
__device__ __forceinline__ float block_sum256(float v, float* red) {
  const int lane = threadIdx.x & 31, wv = threadIdx.x >> 5;
#pragma unroll
  for (int off = 16; off; off >>= 1) v += __shfl_xor(v, off, 32);
  if (lane == 0) red[wv] = v;
  __syncthreads();
  float r = red[lane & 7];
#pragma unroll
  for (int off = 4; off; off >>= 1) r += __shfl_xor(r, off, 32);
  __syncthreads();
  return r;
}

__device__ __forceinline__ int region3(int v, int t0, int t1) {
  return (v < t0) ? 0 : ((v < t1) ? 1 : 2);
}

// --------------------------- kernel 0: weights f32->f16 --------------------
__global__ __launch_bounds__(256) void k_convert_weights(
    const float* __restrict__ qkv_w, const float* __restrict__ proj_w,
    const float* __restrict__ fc1_w, const float* __restrict__ fc2_w,
    _Float16* __restrict__ w16) {
  const int i = blockIdx.x * 256 + threadIdx.x;
  float v;
  if (i < 196608)      v = qkv_w[i];
  else if (i < 262144) v = proj_w[i - 196608];
  else if (i < 524288) v = fc1_w[i - 262144];
  else                 v = fc2_w[i - 524288];
  w16[i] = (_Float16)v;
}

// ------------------- kernel 1: LN1 + cyclic shift + window partition -------
__global__ __launch_bounds__(256) void k_ln1_shift_partition(
    const float* __restrict__ x, const float* __restrict__ g,
    const float* __restrict__ bia, _Float16* __restrict__ xw) {
  __shared__ float red[8];
  const int t  = blockIdx.x;
  const int ww = t % 84;
  const int hh = (t / 84) % 84;
  const int dd = (t / 7056) % 8;
  const int b  = t / 56448;
  const int ds = (dd + 1) & 7;          // roll(-shift)
  const int hs = (hh + 3) % 84;
  const int ws2 = (ww + 3) % 84;
  const int c = threadIdx.x;
  const float v = x[((((size_t)b * 8 + ds) * 256 + c) * 84 + ws2) * 84 + hs];
  const float mu = block_sum256(v, red) * (1.0f / 256.0f);
  const float dv = v - mu;
  const float var = block_sum256(dv * dv, red) * (1.0f / 256.0f);
  const float y = dv * rsqrtf(var + 1e-5f) * g[c] + bia[c];
  const int dwin = dd >> 1, hwin = hh / 7, wwin = ww / 7;
  const int n = ((dd & 1) * 7 + hh % 7) * 7 + ww % 7;
  const int wgl = b * cfg::NWB + (dwin * 12 + hwin) * 12 + wwin;
  xw[((size_t)wgl * 98 + n) * 256 + c] = (_Float16)y;
}

// --------------------------- kernel 2: QKV GEMM ----------------------------
// M=112896, K=256, Nout=768. grid=(7056,6); A strip async-staged in LDS.
__global__ __launch_bounds__(256) void k_qkv_gemm(
    const _Float16* __restrict__ xw, const _Float16* __restrict__ Wq,
    _Float16* __restrict__ qb, _Float16* __restrict__ kbuf,
    _Float16* __restrict__ vbuf) {
  __shared__ _Float16 aslab[16 * 256];
  const int mtile = blockIdx.x;
  async_copy_to_lds(xw + (size_t)mtile * 16 * 256, aslab, 16 * 256 * 2);

  const int wv = threadIdx.x >> 5, lane = threadIdx.x & 31;
  const int ntile = blockIdx.y * 8 + wv;                 // 0..47
  const _Float16* Bw = Wq + (size_t)ntile * 16 * 256;
  v8f acc = {};
  v16h bcur = load_bT_tile(Bw, 256);
#pragma unroll
  for (int k0 = 0; k0 < 256; k0 += 32) {
    v16h bnext = bcur;
    if (k0 + 32 < 256) {
      __builtin_prefetch(Bw + k0 + 96, 0, 3);
      bnext = load_bT_tile(Bw + k0 + 32, 256);
    }
    v16h a = load_a_tile(aslab + k0, 256);
    acc = WMMA16x16x32(a, bcur, acc);
    bcur = bnext;
  }
  const int col = lane & 15;
  const int j = ntile * 16 + col;
  const int which = j >> 8, rem = j & 255, head = rem >> 5, hd = rem & 31;
  _Float16* dst = (which == 0) ? qb : ((which == 1) ? kbuf : vbuf);
  const float sc = (which == 0) ? cfg::SCALE : 1.0f;
  const int rowoff = (lane < 16) ? 0 : 8;
#pragma unroll
  for (int r = 0; r < 8; ++r) {
    const int mrow = mtile * 16 + r + rowoff;
    const int wgl = mrow / 98, n = mrow % 98;
    dst[(((size_t)wgl * 8 + head) * 98 + n) * 32 + hd] = (_Float16)(acc[r] * sc);
  }
}

// --------------------------- kernel 3: windowed attention ------------------
// One block per (window, head); K/V async-staged in LDS; waves 0..6 compute.
__global__ __launch_bounds__(256) void k_attention(
    const _Float16* __restrict__ qb, const _Float16* __restrict__ kbuf,
    const _Float16* __restrict__ vbuf, const float* __restrict__ rpb,
    _Float16* __restrict__ ao) {
  __shared__ _Float16 p_lds[7][16][128];
  __shared__ _Float16 kl[98 * 32];
  __shared__ _Float16 vl[98 * 32];
  const int blk = blockIdx.x;
  const int wgl = blk >> 3, head = blk & 7;
  const size_t base = (((size_t)wgl * 8 + head) * 98) * 32;

  async_copy_to_lds(kbuf + base, kl, 98 * 32 * 2);   // includes barrier
  async_copy_to_lds(vbuf + base, vl, 98 * 32 * 2);

  const int wv = threadIdx.x >> 5, lane = threadIdx.x & 31;
  if (wv >= 7) return;                 // no further cross-wave barriers
  const int mi = wv;
  const int rowoff = (lane < 16) ? 0 : 8;
  const int wIdx = wgl % cfg::NWB;
  const int dwin = wIdx / 144, hwin = (wIdx % 144) / 12, wwin = wIdx % 12;
  const _Float16* Q = qb + base;

  // ---- scores = Q K^T (head_dim 32 == one WMMA K pass per tile) ----
  const v16h aq = load_a_tile_cl(Q, 32, mi * 16, 97);
  v8f s[7];
#pragma unroll
  for (int nt = 0; nt < 7; ++nt) {
    v16h b = load_bT_tile_cl(kl, 32, nt * 16, 97);
    v8f z = {};
    s[nt] = WMMA16x16x32(aq, b, z);
  }

  // ---- + relative position bias + shift mask (computed on the fly) ----
#pragma unroll
  for (int nt = 0; nt < 7; ++nt) {
    const int nk = nt * 16 + (lane & 15);
    const bool kvalid = nk < 98;
    int dk = 0, hk = 0, wk = 0, ck = 0;
    if (kvalid) {
      dk = nk / 49; hk = (nk % 49) / 7; wk = nk % 7;
      ck = region3(dwin * 2 + dk, 6, 7) * 9 +
           region3(hwin * 7 + hk, 77, 81) * 3 +
           region3(wwin * 7 + wk, 77, 81);
    }
#pragma unroll
    for (int r = 0; r < 8; ++r) {
      if (!kvalid) { s[nt][r] = -1e30f; continue; }
      const int nq = mi * 16 + r + rowoff;
      if (nq >= 98) continue;
      const int dq = nq / 49, hq = (nq % 49) / 7, wq = nq % 7;
      const int bi = (((dq - dk + 1) * 13 + (hq - hk + 6)) * 13 + (wq - wk + 6));
      const float bias = rpb[bi * 8 + head];
      const int cq = region3(dwin * 2 + dq, 6, 7) * 9 +
                     region3(hwin * 7 + hq, 77, 81) * 3 +
                     region3(wwin * 7 + wq, 77, 81);
      s[nt][r] += bias + ((cq != ck) ? -100.0f : 0.0f);
    }
  }

  // ---- softmax; each row lives across 16 lanes x 7 tiles ----
#pragma unroll
  for (int r = 0; r < 8; ++r) {
    float m = -1e30f;
#pragma unroll
    for (int nt = 0; nt < 7; ++nt) m = fmaxf(m, s[nt][r]);
#pragma unroll
    for (int off = 8; off; off >>= 1) m = fmaxf(m, __shfl_xor(m, off, 16));
    float sum = 0.0f;
#pragma unroll
    for (int nt = 0; nt < 7; ++nt) {
      const float e = __expf(s[nt][r] - m);
      s[nt][r] = e;
      sum += e;
    }
#pragma unroll
    for (int off = 8; off; off >>= 1) sum += __shfl_xor(sum, off, 16);
    const float inv = 1.0f / sum;
#pragma unroll
    for (int nt = 0; nt < 7; ++nt) s[nt][r] *= inv;
  }

  // ---- D-layout -> A-layout through LDS (per-wave private region) ----
#pragma unroll
  for (int r = 0; r < 8; ++r) {
    const int row = r + rowoff;
#pragma unroll
    for (int nt = 0; nt < 7; ++nt)
      p_lds[mi][row][nt * 16 + (lane & 15)] = (_Float16)s[nt][r];
    p_lds[mi][row][112 + (lane & 15)] = (_Float16)0.0f;   // zero pad K 112..127
  }

  // ---- out = P (16x128, zero-padded) @ V (98x32) ----
#pragma unroll
  for (int hdt = 0; hdt < 2; ++hdt) {
    v8f o = {};
    const int hdcol = hdt * 16 + (lane & 15);
    const int kb2 = (lane < 16) ? 0 : 16;
#pragma unroll
    for (int ck = 0; ck < 4; ++ck) {
      v16h a = load_a_tile(&p_lds[mi][0][ck * 32], 128);
      v16h b;
#pragma unroll
      for (int i = 0; i < 16; ++i) {
        int key = ck * 32 + kb2 + i;
        if (key > 97) key = 97;        // P==0 there; clamp keeps load in-bounds
        b[i] = vl[key * 32 + hdcol];
      }
      o = WMMA16x16x32(a, b, o);
    }
#pragma unroll
    for (int r = 0; r < 8; ++r) {
      const int nq = mi * 16 + r + rowoff;
      if (nq < 98)
        ao[((size_t)wgl * 98 + nq) * 256 + head * 32 + hdcol] = (_Float16)o[r];
    }
  }
}

// -------- kernel 4: proj GEMM + window reverse + un-roll + residual --------
__global__ __launch_bounds__(256) void k_proj_residual(
    const _Float16* __restrict__ ao, const _Float16* __restrict__ Wp,
    const float* __restrict__ proj_b, const float* __restrict__ x,
    float* __restrict__ xres) {
  __shared__ _Float16 aslab[16 * 256];
  const int mtile = blockIdx.x;
  async_copy_to_lds(ao + (size_t)mtile * 16 * 256, aslab, 16 * 256 * 2);

  const int wv = threadIdx.x >> 5, lane = threadIdx.x & 31;
  const int ntile = blockIdx.y * 8 + wv;                 // 0..15
  const _Float16* Bw = Wp + (size_t)ntile * 16 * 256;
  v8f acc = {};
  v16h bcur = load_bT_tile(Bw, 256);
#pragma unroll
  for (int k0 = 0; k0 < 256; k0 += 32) {
    v16h bnext = bcur;
    if (k0 + 32 < 256) bnext = load_bT_tile(Bw + k0 + 32, 256);
    v16h a = load_a_tile(aslab + k0, 256);
    acc = WMMA16x16x32(a, bcur, acc);
    bcur = bnext;
  }
  const int j = ntile * 16 + (lane & 15);
  const float pb = proj_b[j];
  const int rowoff = (lane < 16) ? 0 : 8;
#pragma unroll
  for (int r = 0; r < 8; ++r) {
    const int mrow = mtile * 16 + r + rowoff;
    const int wgl = mrow / 98, n = mrow % 98;
    const int b_ = wgl / cfg::NWB, wIdx = wgl % cfg::NWB;
    const int dwin = wIdx / 144, hwin = (wIdx % 144) / 12, wwin = wIdx % 12;
    const int dq = n / 49, hq = (n % 49) / 7, wq = n % 7;
    const int dp = dwin * 2 + dq, hp = hwin * 7 + hq, wp = wwin * 7 + wq;
    const int d = (dp + 1) & 7, h = (hp + 3) % 84, w = (wp + 3) % 84;  // roll(+s)
    const float sc = x[((((size_t)b_ * 8 + d) * 256 + j) * 84 + w) * 84 + h];
    xres[((((size_t)b_ * 8 + d) * 84 + h) * 84 + w) * 256 + j] = acc[r] + pb + sc;
  }
}

// --------------------------- kernel 5: LN2 ---------------------------------
__global__ __launch_bounds__(256) void k_ln2(
    const float* __restrict__ xres, const float* __restrict__ g,
    const float* __restrict__ bia, _Float16* __restrict__ ln2b) {
  __shared__ float red[8];
  const size_t t = blockIdx.x;
  const int c = threadIdx.x;
  const float v = xres[t * 256 + c];
  const float mu = block_sum256(v, red) * (1.0f / 256.0f);
  const float dv = v - mu;
  const float var = block_sum256(dv * dv, red) * (1.0f / 256.0f);
  ln2b[t * 256 + c] = (_Float16)(dv * rsqrtf(var + 1e-5f) * g[c] + bia[c]);
}

// --------------------------- kernel 6: fc1 GEMM + exact GELU ---------------
__global__ __launch_bounds__(256) void k_fc1_gelu(
    const _Float16* __restrict__ ln2b, const _Float16* __restrict__ W1,
    const float* __restrict__ fc1_b, _Float16* __restrict__ hb) {
  __shared__ _Float16 aslab[16 * 256];
  const int mtile = blockIdx.x;
  async_copy_to_lds(ln2b + (size_t)mtile * 16 * 256, aslab, 16 * 256 * 2);

  const int wv = threadIdx.x >> 5, lane = threadIdx.x & 31;
  const int ntile = blockIdx.y * 8 + wv;                 // 0..63
  const _Float16* Bw = W1 + (size_t)ntile * 16 * 256;
  v8f acc = {};
  v16h bcur = load_bT_tile(Bw, 256);
#pragma unroll
  for (int k0 = 0; k0 < 256; k0 += 32) {
    v16h bnext = bcur;
    if (k0 + 32 < 256) bnext = load_bT_tile(Bw + k0 + 32, 256);
    v16h a = load_a_tile(aslab + k0, 256);
    acc = WMMA16x16x32(a, bcur, acc);
    bcur = bnext;
  }
  const int j = ntile * 16 + (lane & 15);
  const float bj = fc1_b[j];
  const int rowoff = (lane < 16) ? 0 : 8;
#pragma unroll
  for (int r = 0; r < 8; ++r) {
    const int mrow = mtile * 16 + r + rowoff;
    const float tv = acc[r] + bj;
    const float gv = 0.5f * tv * (1.0f + erff(tv * 0.70710678118654752f));
    hb[(size_t)mrow * 1024 + j] = (_Float16)gv;
  }
}

// ------------- kernel 7: fc2 GEMM + residual + output transpose ------------
__global__ __launch_bounds__(256) void k_fc2_out(
    const _Float16* __restrict__ hb, const _Float16* __restrict__ W2,
    const float* __restrict__ fc2_b, const float* __restrict__ xres,
    float* __restrict__ out) {
  __shared__ _Float16 aslab[16 * 1024];
  const int mtile = blockIdx.x;
  async_copy_to_lds(hb + (size_t)mtile * 16 * 1024, aslab, 16 * 1024 * 2);

  const int wv = threadIdx.x >> 5, lane = threadIdx.x & 31;
  const int ntile = blockIdx.y * 8 + wv;                 // 0..15
  const _Float16* Bw = W2 + (size_t)ntile * 16 * 1024;
  v8f acc = {};
  v16h bcur = load_bT_tile(Bw, 1024);
#pragma unroll 4
  for (int k0 = 0; k0 < 1024; k0 += 32) {
    v16h bnext = bcur;
    if (k0 + 32 < 1024) bnext = load_bT_tile(Bw + k0 + 32, 1024);
    v16h a = load_a_tile(aslab + k0, 1024);
    acc = WMMA16x16x32(a, bcur, acc);
    bcur = bnext;
  }
  const int j = ntile * 16 + (lane & 15);
  const float bj = fc2_b[j];
  const int rowoff = (lane < 16) ? 0 : 8;
#pragma unroll
  for (int r = 0; r < 8; ++r) {
    const int token = mtile * 16 + r + rowoff;
    const int w = token % 84;
    const int h = (token / 84) % 84;
    const int d = (token / 7056) % 8;
    const int b_ = token / 56448;
    const float y = acc[r] + bj + xres[(size_t)token * 256 + j];
    out[((((size_t)b_ * 8 + d) * 256 + j) * 84 + w) * 84 + h] = y;
  }
}

// ------------------------------- launcher ----------------------------------
extern "C" void kernel_launch(void* const* d_in, const int* in_sizes, int n_in,
                              void* d_out, int out_size, void* d_ws, size_t ws_size,
                              hipStream_t stream) {
  const float* x       = (const float*)d_in[0];
  const float* norm1_g = (const float*)d_in[1];
  const float* norm1_b = (const float*)d_in[2];
  const float* qkv_w   = (const float*)d_in[3];
  const float* rpb     = (const float*)d_in[4];
  const float* proj_w  = (const float*)d_in[5];
  const float* proj_b  = (const float*)d_in[6];
  const float* norm2_g = (const float*)d_in[7];
  const float* norm2_b = (const float*)d_in[8];
  const float* fc1_w   = (const float*)d_in[9];
  const float* fc1_b   = (const float*)d_in[10];
  const float* fc2_w   = (const float*)d_in[11];
  const float* fc2_b   = (const float*)d_in[12];
  float* out = (float*)d_out;

  // workspace layout (~695 MB total)
  const size_t TC = (size_t)cfg::T * 256;   // 28,901,376
  _Float16* w16  = (_Float16*)d_ws;                       // 786432 f16
  _Float16* xw   = (_Float16*)((char*)d_ws + 786432 * 2);
  _Float16* qb   = xw + TC;
  _Float16* kb   = qb + TC;
  _Float16* vb   = kb + TC;
  _Float16* ao   = vb + TC;
  float*    xres = (float*)(ao + TC);
  _Float16* ln2b = (_Float16*)(xres + TC);
  _Float16* hb   = ln2b + TC;                             // T*1024 f16

  _Float16* qkvW  = w16;
  _Float16* projW = w16 + 196608;
  _Float16* fc1W  = w16 + 262144;
  _Float16* fc2W  = w16 + 524288;

  k_convert_weights<<<3072, 256, 0, stream>>>(qkv_w, proj_w, fc1_w, fc2_w, w16);
  k_ln1_shift_partition<<<cfg::T, 256, 0, stream>>>(x, norm1_g, norm1_b, xw);
  k_qkv_gemm<<<dim3(cfg::MT, 6), 256, 0, stream>>>(xw, qkvW, qb, kb, vb);
  k_attention<<<cfg::BW * cfg::NH, 256, 0, stream>>>(qb, kb, vb, rpb, ao);
  k_proj_residual<<<dim3(cfg::MT, 2), 256, 0, stream>>>(ao, projW, proj_b, x, xres);
  k_ln2<<<cfg::T, 256, 0, stream>>>(xres, norm2_g, norm2_b, ln2b);
  k_fc1_gelu<<<dim3(cfg::MT, 8), 256, 0, stream>>>(ln2b, fc1W, fc1_b, hb);
  k_fc2_out<<<dim3(cfg::MT, 2), 256, 0, stream>>>(hb, fc2W, fc2_b, xres, out);
}
